// PMA_90443421319312
// MI455X (gfx1250) — matrix operand order repaired and checked by reading, not compiled, not run on hardware
//
#include <hip/hip_runtime.h>
#include <hip/hip_bf16.h>

typedef __attribute__((ext_vector_type(8)))  float   v8f;
typedef __attribute__((ext_vector_type(8)))  __bf16  v8bf;
typedef __attribute__((ext_vector_type(16))) __bf16  v16bf;

#define DIMX  4096
#define DVX   1024
#define HX    16
#define DH    64
#define NSX   32
#define BX    8
#define LX    2048
#define LTILE 128
#define BK    64

// D = A(16x32 bf16) * B(32x16 bf16) + C(16x16 f32)
__device__ __forceinline__ v8f wmma_bf16(v16bf a, v16bf b, v8f c) {
  return __builtin_amdgcn_wmma_f32_16x16x32_bf16(false, a, false, b, (short)0, c, false, false);
}

// Async global->LDS copy, 16B per lane, GVS mode (SGPR base + VGPR offset).
// Tracked by ASYNCcnt; LDS dest address carried in a VGPR (no data VGPRs).
__device__ __forceinline__ void async_b128(unsigned lds, unsigned goff,
                                           unsigned long long base) {
  asm volatile("global_load_async_to_lds_b128 %0, %1, %2"
               :: "v"(lds), "v"(goff), "s"(base) : "memory");
}
__device__ __forceinline__ void wait_async0() {
  asm volatile("s_wait_asynccnt 0" ::: "memory");
}
__device__ __forceinline__ unsigned lds_off(const void* p) {
  return (unsigned)(size_t)p; // LDS aperture: low 32 bits = wave LDS offset
}

// A-frag k order used here (contiguous, B reads match): lanes 0-15 hold M=lane, K={0..15};
// lanes 16-31 hold M=lane-16, K={16..31}.
__device__ __forceinline__ v16bf make_afrag(const __bf16* rowp, int lhi) {
  const int o = lhi ? 16 : 0;
  v8bf lo = *(const v8bf*)(rowp + o);
  v8bf hi = *(const v8bf*)(rowp + 8 + o);
  v16bf a;
#pragma unroll
  for (int i = 0; i < 8; ++i) { a[i] = lo[i]; a[8 + i] = hi[i]; }
  return a;
}

// Same A-frag but sourced from f32 LDS (async-staged X), cvt to bf16 inline;
// v_cvt_pk_bf16_f32 co-executes with the XDL WMMA pipe.
__device__ __forceinline__ v16bf make_afrag_f32(const float* rowp, int lhi) {
  const int o = lhi ? 16 : 0;
  float4 l0 = *(const float4*)(rowp + o);
  float4 l1 = *(const float4*)(rowp + o + 4);
  float4 h0 = *(const float4*)(rowp + 8 + o);
  float4 h1 = *(const float4*)(rowp + 8 + o + 4);
  v16bf a;
  a[0] = (__bf16)l0.x; a[1] = (__bf16)l0.y; a[2]  = (__bf16)l0.z; a[3]  = (__bf16)l0.w;
  a[4] = (__bf16)l1.x; a[5] = (__bf16)l1.y; a[6]  = (__bf16)l1.z; a[7]  = (__bf16)l1.w;
  a[8] = (__bf16)h0.x; a[9] = (__bf16)h0.y; a[10] = (__bf16)h0.z; a[11] = (__bf16)h0.w;
  a[12] = (__bf16)h1.x; a[13] = (__bf16)h1.y; a[14] = (__bf16)h1.z; a[15] = (__bf16)h1.w;
  return a;
}

// ---------------------------------------------------------------- weights cvt
__global__ void __launch_bounds__(256) cvt_weights(
    const float* __restrict__ Wk, const float* __restrict__ Wv,
    __bf16* __restrict__ WkB, __bf16* __restrict__ WvB) {
  size_t i = ((size_t)blockIdx.x * 256 + threadIdx.x) * 8;
  float4 a0 = *(const float4*)(Wk + i);
  float4 a1 = *(const float4*)(Wk + i + 4);
  v8bf ok;
  ok[0] = (__bf16)a0.x; ok[1] = (__bf16)a0.y; ok[2] = (__bf16)a0.z; ok[3] = (__bf16)a0.w;
  ok[4] = (__bf16)a1.x; ok[5] = (__bf16)a1.y; ok[6] = (__bf16)a1.z; ok[7] = (__bf16)a1.w;
  *(v8bf*)(WkB + i) = ok;
  float4 b0 = *(const float4*)(Wv + i);
  float4 b1 = *(const float4*)(Wv + i + 4);
  v8bf ov;
  ov[0] = (__bf16)b0.x; ov[1] = (__bf16)b0.y; ov[2] = (__bf16)b0.z; ov[3] = (__bf16)b0.w;
  ov[4] = (__bf16)b1.x; ov[5] = (__bf16)b1.y; ov[6] = (__bf16)b1.z; ov[7] = (__bf16)b1.w;
  *(v8bf*)(WvB + i) = ov;
}

// ---------------------------------------------------------------- Q projection
__global__ void __launch_bounds__(256) q_proj(
    const float* __restrict__ S, const float* __restrict__ Wq,
    const float* __restrict__ bq, __bf16* __restrict__ Qpb) {
  const int q = blockIdx.x;
  const int tid = threadIdx.x;
  __shared__ float srow[DVX];
#pragma unroll
  for (int i = 0; i < 4; ++i) srow[tid + i * 256] = S[(size_t)q * DVX + tid + i * 256];
  __syncthreads();
#pragma unroll
  for (int ii = 0; ii < 4; ++ii) {
    int j = tid + ii * 256;
    const float4* w4 = (const float4*)(Wq + (size_t)j * DVX);
    float acc = bq[j];
    for (int k = 0; k < DVX / 4; ++k) {
      float4 w = w4[k];
      float4 s4 = *(const float4*)&srow[k * 4];
      acc += w.x * s4.x + w.y * s4.y + w.z * s4.z + w.w * s4.w;
    }
    Qpb[(size_t)q * DVX + j] = (__bf16)(acc * 0.03125f); // fold 1/sqrt(1024)
  }
}

// ---------------------------------------------------------------- K/V proj GEMM
// K = bf16(X@Wk^T + bk), V = bf16(X@Wv^T + bv).  Block tile 64M x 128N x 64K.
// Double-buffered LDS, all staging via async global->LDS b128 (ASYNCcnt).
__global__ void __launch_bounds__(256) kv_proj(
    const float* __restrict__ X, const __bf16* __restrict__ WkB,
    const __bf16* __restrict__ WvB, const float* __restrict__ bk,
    const float* __restrict__ bv, __bf16* __restrict__ Kb, __bf16* __restrict__ Vb) {
  const int bm = blockIdx.x * 64;
  const int bn = blockIdx.y * 128;
  const int tid = threadIdx.x;
  const int wave = tid >> 5, lane = tid & 31;
  const int lhi = lane >> 4, lm = lane & 15;
  const int wm = wave & 3, wn = wave >> 2; // 4x2 wave grid, 16M x 64N per wave

  __shared__ __attribute__((aligned(32))) float  Xs[2][64][BK];    // 32 KB (f32)
  __shared__ __attribute__((aligned(32))) __bf16 Wks[2][128][BK];  // 32 KB
  __shared__ __attribute__((aligned(32))) __bf16 Wvs[2][128][BK];  // 32 KB
  __shared__ __attribute__((aligned(32))) __bf16 Cs[64][128];      // 16 KB

  // staging assignments (per thread): X 64x64 f32 -> 4x b128;  W 128x64 bf16 -> 4x b128 each
  const int rX = tid >> 2, cX = (tid & 3) * 16;
  const int rW = tid >> 1, cW = (tid & 1) * 32;
  unsigned ldsX[2]  = { lds_off(&Xs[0][rX][cX]),  lds_off(&Xs[1][rX][cX])  };
  unsigned ldsWk[2] = { lds_off(&Wks[0][rW][cW]), lds_off(&Wks[1][rW][cW]) };
  unsigned ldsWv[2] = { lds_off(&Wvs[0][rW][cW]), lds_off(&Wvs[1][rW][cW]) };
  const unsigned goffX = (unsigned)(((size_t)(bm + rX) * DIMX + cX) * 4);
  const unsigned goffW = (unsigned)(((size_t)(bn + rW) * DIMX + cW) * 2);
  const unsigned long long Xb = (unsigned long long)X;
  const unsigned long long Wkb = (unsigned long long)WkB;
  const unsigned long long Wvb = (unsigned long long)WvB;

  auto issue_tile = [&](int p, int k0) {
    unsigned gx = goffX + (unsigned)k0 * 4;
#pragma unroll
    for (int i = 0; i < 4; ++i) async_b128(ldsX[p] + i * 16, gx + i * 16, Xb);
    unsigned gw = goffW + (unsigned)k0 * 2;
#pragma unroll
    for (int i = 0; i < 4; ++i) async_b128(ldsWk[p] + i * 16, gw + i * 16, Wkb);
#pragma unroll
    for (int i = 0; i < 4; ++i) async_b128(ldsWv[p] + i * 16, gw + i * 16, Wvb);
  };

  v8f accK[4], accV[4];
#pragma unroll
  for (int t = 0; t < 4; ++t) { accK[t] = {}; accV[t] = {}; }

  issue_tile(0, 0);
  for (int it = 0; it < DIMX / BK; ++it) {
    const int p = it & 1;
    wait_async0();       // own async copies for buffer p complete
    __syncthreads();     // all waves' copies complete; prior reads of 1-p done
    if (it + 1 < DIMX / BK) issue_tile(1 - p, (it + 1) * BK);

    const float* xrow = &Xs[p][wm * 16 + lm][0];
    v16bf a0 = make_afrag_f32(xrow, lhi);
    v16bf a1 = make_afrag_f32(xrow + 32, lhi);
#pragma unroll
    for (int t = 0; t < 4; ++t) {
      int n = wn * 64 + t * 16 + lm;
      v16bf bK0 = *(const v16bf*)&Wks[p][n][lhi * 16];
      v16bf bK1 = *(const v16bf*)&Wks[p][n][32 + lhi * 16];
      v16bf bV0 = *(const v16bf*)&Wvs[p][n][lhi * 16];
      v16bf bV1 = *(const v16bf*)&Wvs[p][n][32 + lhi * 16];
      accK[t] = wmma_bf16(a0, bK0, accK[t]);
      accK[t] = wmma_bf16(a1, bK1, accK[t]);
      accV[t] = wmma_bf16(a0, bV0, accV[t]);
      accV[t] = wmma_bf16(a1, bV1, accV[t]);
    }
    __syncthreads();     // done reading buffer p
  }

  // epilogue: bias + bf16, staged through LDS for coalesced b128 stores
#pragma unroll
  for (int t = 0; t < 4; ++t) {
    int col = wn * 64 + t * 16 + lm;
    float bias = bk[bn + col];
#pragma unroll
    for (int i = 0; i < 8; ++i)
      Cs[wm * 16 + (lhi ? 8 : 0) + i][col] = (__bf16)(accK[t][i] + bias);
  }
  __syncthreads();
  {
    int r = tid >> 2, c0 = (tid & 3) * 32;
    __bf16* dst = Kb + (size_t)(bm + r) * DVX + bn + c0;
#pragma unroll
    for (int i = 0; i < 4; ++i) *(v8bf*)(dst + i * 8) = *(const v8bf*)&Cs[r][c0 + i * 8];
  }
  __syncthreads();
#pragma unroll
  for (int t = 0; t < 4; ++t) {
    int col = wn * 64 + t * 16 + lm;
    float bias = bv[bn + col];
#pragma unroll
    for (int i = 0; i < 8; ++i)
      Cs[wm * 16 + (lhi ? 8 : 0) + i][col] = (__bf16)(accV[t][i] + bias);
  }
  __syncthreads();
  {
    int r = tid >> 2, c0 = (tid & 3) * 32;
    __bf16* dst = Vb + (size_t)(bm + r) * DVX + bn + c0;
#pragma unroll
    for (int i = 0; i < 4; ++i) *(v8bf*)(dst + i * 8) = *(const v8bf*)&Cs[r][c0 + i * 8];
  }
}

// ---------------------------------------------------------------- attention
// One block per (b,h). Flash-style online softmax over L in 128-key tiles.
__global__ void __launch_bounds__(256) pma_attn(
    const __bf16* __restrict__ Qpb, const __bf16* __restrict__ Kb,
    const __bf16* __restrict__ Vb, const int* __restrict__ pm,
    float* __restrict__ Oattn) {
  const int b = blockIdx.x >> 4;
  const int h = blockIdx.x & 15;
  const int tid = threadIdx.x;
  const int wave = tid >> 5, lane = tid & 31;
  const int lhi = lane >> 4, lm = lane & 15;
  const int mq = wave & 1, nd = wave >> 1;

  __shared__ __attribute__((aligned(32))) __bf16 Qs[32][DH];
  __shared__ __attribute__((aligned(32))) __bf16 Ks[LTILE][DH];
  __shared__ __attribute__((aligned(32))) __bf16 Vt[DH][LTILE + 8];
  __shared__ __attribute__((aligned(32))) float  Ss[32][LTILE];
  __shared__ __attribute__((aligned(32))) __bf16 Ps[32][LTILE];
  __shared__ float mrow[32], crow[32], lrow[32];
  __shared__ int msk[LTILE];

  { // Q head slice -> LDS
    int r = tid >> 3, c = (tid & 7) * 8;
    *(v8bf*)&Qs[r][c] = *(const v8bf*)(Qpb + (size_t)r * DVX + h * DH + c);
  }
  if (tid < 32) { mrow[tid] = -1e30f; lrow[tid] = 0.f; }
  __syncthreads();

  v16bf aq[2][2];
#pragma unroll
  for (int m = 0; m < 2; ++m) {
    const __bf16* rowp = &Qs[m * 16 + lm][0];
    aq[m][0] = make_afrag(rowp, lhi);
    aq[m][1] = make_afrag(rowp + 32, lhi);
  }

  const int rT = tid >> 1, cT = (tid & 1) * 32;
  const unsigned ldsK = lds_off(&Ks[rT][cT]);
  const unsigned long long Kbase = (unsigned long long)Kb;

  v8f acc = {};

  for (int l0 = 0; l0 < LX; l0 += LTILE) {
    { // stage K via async-to-LDS; V manually (transpose scatter); mask scalar
      unsigned goff = (unsigned)(((size_t)(b * LX + l0 + rT) * DVX + h * DH + cT) * 2);
#pragma unroll
      for (int i = 0; i < 4; ++i) async_b128(ldsK + i * 16, goff + i * 16, Kbase);
      const __bf16* vp = Vb + (size_t)(b * LX + l0 + rT) * DVX + h * DH + cT;
#pragma unroll
      for (int i = 0; i < 4; ++i) {
        v8bf v = *(const v8bf*)(vp + i * 8);
#pragma unroll
        for (int j = 0; j < 8; ++j) Vt[cT + i * 8 + j][rT] = v[j];
      }
      if (tid < LTILE) msk[tid] = pm[(size_t)b * LX + l0 + tid] != 0;
    }
    wait_async0();
    __syncthreads();

    { // scores: wave w owns key-subtile w (16 keys), both q-subtiles
      const __bf16* krow = &Ks[wave * 16 + lm][0];
      v16bf bk0 = *(const v16bf*)(krow + lhi * 16);
      v16bf bk1 = *(const v16bf*)(krow + 32 + lhi * 16);
      v8f sc0 = {}, sc1 = {};
      sc0 = wmma_bf16(aq[0][0], bk0, sc0);
      sc0 = wmma_bf16(aq[0][1], bk1, sc0);
      sc1 = wmma_bf16(aq[1][0], bk0, sc1);
      sc1 = wmma_bf16(aq[1][1], bk1, sc1);
      int n = wave * 16 + lm;
#pragma unroll
      for (int i = 0; i < 8; ++i) {
        Ss[(lhi ? 8 : 0) + i][n] = sc0[i];
        Ss[16 + (lhi ? 8 : 0) + i][n] = sc1[i];
      }
    }
    __syncthreads();

    if (tid < 32) { // online softmax row pass
      const int q = tid;
      float mold = mrow[q], mt = mold;
      for (int k = 0; k < LTILE; ++k)
        if (msk[k]) mt = fmaxf(mt, Ss[q][k]);
      float c = __expf(mold - mt);
      float lsum = lrow[q] * c;
      for (int k = 0; k < LTILE; ++k) {
        float p = msk[k] ? __expf(Ss[q][k] - mt) : 0.f;
        Ps[q][k] = (__bf16)p;
        lsum += p;
      }
      mrow[q] = mt; crow[q] = c; lrow[q] = lsum;
    }
    __syncthreads();

    { // rescale accumulator, then acc += P @ V
      const int qb = mq * 16 + (lhi ? 8 : 0);
#pragma unroll
      for (int i = 0; i < 8; ++i) acc[i] *= crow[qb + i];
    }
    {
      const __bf16* prow = &Ps[mq * 16 + lm][0];
      const __bf16* vrow = &Vt[nd * 16 + lm][0];
#pragma unroll
      for (int kc = 0; kc < 4; ++kc) {
        v16bf ap  = make_afrag(prow + kc * 32, lhi);
        v16bf bvf = *(const v16bf*)(vrow + kc * 32 + lhi * 16);
        acc = wmma_bf16(ap, bvf, acc);
      }
    }
    __syncthreads();
  }

  { // final: divide by row sum (0 if row fully masked)
    const int qb = mq * 16 + (lhi ? 8 : 0);
    const int d = h * DH + nd * 16 + lm;
#pragma unroll
    for (int i = 0; i < 8; ++i) {
      float l = lrow[qb + i];
      Oattn[(size_t)(b * NSX + qb + i) * DVX + d] = (l > 0.f) ? acc[i] / l : 0.f;
    }
  }
}

// ---------------------------------------------------------------- LN helpers
__device__ __forceinline__ void block_sum2(float& s0, float& s1, float* red) {
  const int tid = threadIdx.x;
  red[tid] = s0; red[256 + tid] = s1;
  __syncthreads();
  for (int off = 128; off > 0; off >>= 1) {
    if (tid < off) { red[tid] += red[tid + off]; red[256 + tid] += red[256 + tid + off]; }
    __syncthreads();
  }
  s0 = red[0]; s1 = red[256];
  __syncthreads();
}

__global__ void __launch_bounds__(256) resid_ln0(
    const float* __restrict__ Oattn, const float* __restrict__ S,
    const float* __restrict__ g0, const float* __restrict__ be0,
    float* __restrict__ T) {
  const int row = blockIdx.x;
  const int q = row & (NSX - 1);
  const int tid = threadIdx.x;
  __shared__ float red[512];
  float v[4], s = 0.f, s2 = 0.f;
#pragma unroll
  for (int i = 0; i < 4; ++i) {
    int j = tid + i * 256;
    float x = Oattn[(size_t)row * DVX + j] + S[(size_t)q * DVX + j];
    v[i] = x; s += x; s2 += x * x;
  }
  block_sum2(s, s2, red);
  float mu = s * (1.f / DVX);
  float var = s2 * (1.f / DVX) - mu * mu;
  float inv = rsqrtf(var + 1e-5f);
#pragma unroll
  for (int i = 0; i < 4; ++i) {
    int j = tid + i * 256;
    T[(size_t)row * DVX + j] = (v[i] - mu) * inv * g0[j] + be0[j];
  }
}

__global__ void __launch_bounds__(256) ffn_ln1(
    const float* __restrict__ T, const float* __restrict__ Wo,
    const float* __restrict__ bo, const float* __restrict__ g1,
    const float* __restrict__ be1, float* __restrict__ Out) {
  const int row = blockIdx.x;
  const int tid = threadIdx.x;
  __shared__ float tl[DVX];
  __shared__ float u[DVX];
  __shared__ float red[512];
#pragma unroll
  for (int i = 0; i < 4; ++i) tl[tid + i * 256] = T[(size_t)row * DVX + tid + i * 256];
  __syncthreads();
#pragma unroll
  for (int ii = 0; ii < 4; ++ii) {
    int j = tid + ii * 256;
    const float4* w4 = (const float4*)(Wo + (size_t)j * DVX);
    float acc = bo[j];
    for (int k = 0; k < DVX / 4; ++k) {
      float4 w = w4[k];
      float4 t4 = *(const float4*)&tl[k * 4];
      acc += w.x * t4.x + w.y * t4.y + w.z * t4.z + w.w * t4.w;
    }
    u[j] = tl[j] + (acc > 0.f ? acc : 0.f);
  }
  __syncthreads();
  float v[4], s = 0.f, s2 = 0.f;
#pragma unroll
  for (int i = 0; i < 4; ++i) {
    int j = tid + i * 256;
    float x = u[j];
    v[i] = x; s += x; s2 += x * x;
  }
  block_sum2(s, s2, red);
  float mu = s * (1.f / DVX);
  float var = s2 * (1.f / DVX) - mu * mu;
  float inv = rsqrtf(var + 1e-5f);
#pragma unroll
  for (int i = 0; i < 4; ++i) {
    int j = tid + i * 256;
    Out[(size_t)row * DVX + j] = (v[i] - mu) * inv * g1[j] + be1[j];
  }
}

// ---------------------------------------------------------------- launch
extern "C" void kernel_launch(void* const* d_in, const int* in_sizes, int n_in,
                              void* d_out, int out_size, void* d_ws, size_t ws_size,
                              hipStream_t stream) {
  const float* X   = (const float*)d_in[0];
  const int*   pm  = (const int*)d_in[1];
  const float* S   = (const float*)d_in[2];
  const float* Wq  = (const float*)d_in[3];
  const float* bq  = (const float*)d_in[4];
  const float* Wk  = (const float*)d_in[5];
  const float* bk  = (const float*)d_in[6];
  const float* Wv  = (const float*)d_in[7];
  const float* bv  = (const float*)d_in[8];
  const float* Wo  = (const float*)d_in[9];
  const float* bo  = (const float*)d_in[10];
  const float* g0  = (const float*)d_in[11];
  const float* be0 = (const float*)d_in[12];
  const float* g1  = (const float*)d_in[13];
  const float* be1 = (const float*)d_in[14];
  float* Out = (float*)d_out;

  char* ws = (char*)d_ws;
  size_t o = 0;
  __bf16* WkB  = (__bf16*)(ws + o); o += (size_t)DVX * DIMX * 2;
  __bf16* WvB  = (__bf16*)(ws + o); o += (size_t)DVX * DIMX * 2;
  __bf16* Kb   = (__bf16*)(ws + o); o += (size_t)BX * LX * DVX * 2;
  __bf16* Vb   = (__bf16*)(ws + o); o += (size_t)BX * LX * DVX * 2;
  __bf16* Qpb  = (__bf16*)(ws + o); o += (size_t)NSX * DVX * 2;
  float* Oattn = (float*)(ws + o);  o += (size_t)BX * NSX * DVX * 4;
  float* Tbuf  = (float*)(ws + o);  o += (size_t)BX * NSX * DVX * 4;

  cvt_weights<<<2048, 256, 0, stream>>>(Wk, Wv, WkB, WvB);
  q_proj<<<NSX, 256, 0, stream>>>(S, Wq, bq, Qpb);
  kv_proj<<<dim3(256, 8), 256, 0, stream>>>(X, WkB, WvB, bk, bv, Kb, Vb);
  pma_attn<<<BX * HX, 256, 0, stream>>>(Qpb, Kb, Vb, pm, Oattn);
  resid_ln0<<<BX * NSX, 256, 0, stream>>>(Oattn, S, g0, be0, Tbuf);
  ffn_ln1<<<BX * NSX, 256, 0, stream>>>(Tbuf, Wo, bo, g1, be1, Out);
}
